// FusedMoEBlock_27676769255764
// MI455X (gfx1250) — matrix operand order repaired and checked by reading, not compiled, not run on hardware
//
#include <hip/hip_runtime.h>
#include <hip/hip_bf16.h>

// ---------------------------------------------------------------------------
// Fused MoE block for MI455X (gfx1250, wave32, WMMA f32_16x16x32_f16)
// ---------------------------------------------------------------------------

typedef __attribute__((ext_vector_type(16))) _Float16 v16h;
typedef __attribute__((ext_vector_type(8)))  float    v8f;
typedef __attribute__((__vector_size__(16))) int      v4i_raw;

// Problem constants (from reference setup_inputs)
#define TT   2048        // tokens = B*S
#define HH   2048        // hidden
#define EE   64          // experts
#define II   512         // expert intermediate
#define ISS  1024        // shared intermediate
#define TOPK 6
#define NROWS (TT * TOPK)   // 12288 gathered expert rows (exact)

__device__ __forceinline__ float fast_rcp(float x) {
    return __builtin_amdgcn_rcpf(x);
}
__device__ __forceinline__ float silu_f(float g) {
    return g * fast_rcp(1.f + __expf(-g));
}

#define WMMA_F16(a, b, c) \
    __builtin_amdgcn_wmma_f32_16x16x32_f16(false, (a), false, (b), (short)0, (c), false, false)

// ---------------------------------------------------------------------------
// Async global->LDS copy (CDNA5 GLOBAL_LOAD_ASYNC_TO_LDS_B128, ASYNCcnt)
// with a safe synchronous fallback when the builtin is unavailable.
// Builtin signature (from compiler diagnostic): non-const v4i* in AS1 / AS3.
// ---------------------------------------------------------------------------
#if __has_builtin(__builtin_amdgcn_global_load_async_to_lds_b128)
#define MOE_ASYNC_LDS 1
#endif

__device__ __forceinline__ void copy16_g2l(const _Float16* __restrict__ g,
                                           _Float16* l) {
#ifdef MOE_ASYNC_LDS
    __builtin_amdgcn_global_load_async_to_lds_b128(
        (__attribute__((address_space(1))) v4i_raw*)g,
        (__attribute__((address_space(3))) v4i_raw*)l, 0, 0);
#else
    *(uint4*)l = *(const uint4*)g;
#endif
}

__device__ __forceinline__ void async_copy_wait() {
#ifdef MOE_ASYNC_LDS
#if __has_builtin(__builtin_amdgcn_s_wait_asynccnt)
    __builtin_amdgcn_s_wait_asynccnt(0);
#else
    asm volatile("s_wait_asynccnt 0x0" ::: "memory");
#endif
#endif
}

// ---------------------------------------------------------------------------
// 1) f32 -> f16 conversion of activations (vectorized: 4 elems/thread)
// ---------------------------------------------------------------------------
__global__ void cvt_f32_f16_kernel(const float* __restrict__ in,
                                   _Float16* __restrict__ out, int n4) {
    int i = blockIdx.x * blockDim.x + threadIdx.x;
    if (i >= n4) return;
    float4 v = ((const float4*)in)[i];
    union { _Float16 h[4]; uint2 u; } o;
    o.h[0] = (_Float16)v.x; o.h[1] = (_Float16)v.y;
    o.h[2] = (_Float16)v.z; o.h[3] = (_Float16)v.w;
    ((uint2*)out)[i] = o.u;
}

// ---------------------------------------------------------------------------
// 2) Router: logits -> sigmoid -> normalize -> top-6 -> renormalize
//    one block per token, 64 threads (one per expert)
// ---------------------------------------------------------------------------
__global__ void router_kernel(const float* __restrict__ x,
                              const float* __restrict__ gw,
                              int* __restrict__ topk_idx,
                              float* __restrict__ topk_w) {
    __shared__ float xs[HH];
    __shared__ float ps[EE];
    __shared__ float red[EE];
    __shared__ int   redi[EE];

    const int t = blockIdx.x;
    const int e = threadIdx.x;

    for (int i = e; i < HH; i += EE) xs[i] = x[(size_t)t * HH + i];
    __syncthreads();

    const float* w = gw + (size_t)e * HH;
    float acc = 0.f;
    for (int i = 0; i < HH; ++i) acc += xs[i] * w[i];

    float p = fast_rcp(1.f + __expf(-acc));
    red[e] = p;
    __syncthreads();
    for (int s = 32; s > 0; s >>= 1) {
        if (e < s) red[e] += red[e + s];
        __syncthreads();
    }
    const float psum = red[0];
    __syncthreads();
    ps[e] = p * fast_rcp(psum);
    __syncthreads();

    for (int k = 0; k < TOPK; ++k) {
        red[e] = ps[e]; redi[e] = e;
        __syncthreads();
        for (int s = 32; s > 0; s >>= 1) {
            if (e < s) {
                float ov = red[e + s]; int oi = redi[e + s];
                if (ov > red[e] || (ov == red[e] && oi < redi[e])) {
                    red[e] = ov; redi[e] = oi;
                }
            }
            __syncthreads();
        }
        if (e == 0) {
            topk_idx[t * TOPK + k] = redi[0];
            topk_w [t * TOPK + k] = red[0];
            ps[redi[0]] = -1.f;
        }
        __syncthreads();
    }
    if (e == 0) {
        float s = 0.f;
        for (int k = 0; k < TOPK; ++k) s += topk_w[t * TOPK + k];
        float inv = fast_rcp(s);
        for (int k = 0; k < TOPK; ++k) topk_w[t * TOPK + k] *= inv;
    }
}

// ---------------------------------------------------------------------------
// 3) Deterministic per-expert token list build (one block, 64 threads)
// ---------------------------------------------------------------------------
__global__ void build_lists_kernel(const int* __restrict__ topk_idx,
                                   int* __restrict__ counts,
                                   int* __restrict__ offs,
                                   int* __restrict__ ltok,
                                   int* __restrict__ row_of) {
    __shared__ int c[EE];
    __shared__ int o[EE + 1];
    const int e = threadIdx.x;

    int cnt = 0;
    for (int t = 0; t < TT; ++t)
        for (int k = 0; k < TOPK; ++k)
            if (topk_idx[t * TOPK + k] == e) ++cnt;
    c[e] = cnt;
    __syncthreads();
    if (e == 0) {
        o[0] = 0;
        for (int i = 0; i < EE; ++i) o[i + 1] = o[i] + c[i];
    }
    __syncthreads();
    const int base = o[e];
    counts[e] = cnt;
    offs[e]   = base;
    int n = 0;
    for (int t = 0; t < TT; ++t)
        for (int k = 0; k < TOPK; ++k)
            if (topk_idx[t * TOPK + k] == e) {
                int r = base + n;
                ltok[r] = t;
                row_of[t * TOPK + k] = r;
                ++n;
            }
}

// ---------------------------------------------------------------------------
// WMMA GEMM helpers
//   A tile: 32(M) x 32(K) f16, row-major in LDS          (async b128 copies)
//   B tile: 32(K) x 128(N) f32 -> f16 transposed [n][k]  (b32 packed stores)
//   Fragments: A and B both use contiguous k = 16*half + s (lane half, element s).
// ---------------------------------------------------------------------------
__device__ __forceinline__ void load_a_frag(const _Float16* __restrict__ As,
                                            int lane, v16h& a) {
    const int half = lane >> 4;
    const _Float16* ar = &As[(lane & 15) * 32];
    const int kb0 = half * 16;
    const int kb1 = half * 16 + 8;
#pragma unroll
    for (int j = 0; j < 8; ++j) a[j]     = ar[kb0 + j];
#pragma unroll
    for (int j = 0; j < 8; ++j) a[8 + j] = ar[kb1 + j];
}

__device__ __forceinline__ void load_b_frag(const _Float16* __restrict__ Bs,
                                            int nrow, int lane, v16h& b) {
    const int kb = (lane >> 4) * 16;
    const _Float16* br = &Bs[nrow * 32 + kb];
#pragma unroll
    for (int j = 0; j < 16; ++j) b[j] = br[j];
}

// Stage one 32x128 f32 weight tile into LDS as transposed f16 [n][k].
// 128 threads: each owns 2 consecutive K rows x 16 N cols -> packed b32 stores.
__device__ __forceinline__ void stage_b_tile(const float* __restrict__ src,
                                             int ndim, _Float16* __restrict__ Bs,
                                             int tid) {
    const int k  = (tid & 15) * 2;      // 0,2,...,30
    const int n  = (tid >> 4) * 16;     // 0,16,...,112
    const float* r0 = src + (size_t)k * ndim + n;
    const float* r1 = r0 + ndim;
#pragma unroll
    for (int j = 0; j < 16; ++j) {
        union { _Float16 h[2]; unsigned int u; } pk;
        pk.h[0] = (_Float16)r0[j];
        pk.h[1] = (_Float16)r1[j];
        *(unsigned int*)&Bs[(n + j) * 32 + k] = pk.u;
    }
}

// ---------------------------------------------------------------------------
// 4) Gated (gate/up + SiLU) WMMA GEMM.  Block tile M=32 x N=128, K-step 32.
//    4 waves; each wave owns 2 M-subtiles x 2 N-subtiles -> 8 WMMA / K-step.
// ---------------------------------------------------------------------------
template<bool GATHER, int KDIM, int NDIM>
__global__ void gateup_kernel(const _Float16* __restrict__ A,   // [*, KDIM] f16
                              const float* __restrict__ Wg,     // [e][KDIM][NDIM]
                              const float* __restrict__ Wu,
                              const int* __restrict__ counts,
                              const int* __restrict__ offs,
                              const int* __restrict__ ltok,
                              _Float16* __restrict__ Out) {     // [rows][NDIM]
    const int e   = blockIdx.z;
    const int cnt = GATHER ? counts[e] : TT;
    const int m0  = blockIdx.y * 32;
    if (m0 >= cnt) return;
    const int base = GATHER ? offs[e] : 0;
    const int n0   = blockIdx.x * 128;

    const size_t wsz = (size_t)KDIM * NDIM;
    const float* wg = Wg + (size_t)e * wsz + n0;
    const float* wu = Wu + (size_t)e * wsz + n0;

    __shared__ _Float16 As[32 * 32];
    __shared__ _Float16 Bg[128 * 32];   // transposed [n][k]
    __shared__ _Float16 Bu[128 * 32];

    const int tid  = threadIdx.x;
    const int lane = tid & 31;
    const int wave = tid >> 5;
    const int half = lane >> 4;

    // Hoisted, loop-invariant A-tile addressing (gather resolved once).
    const int ar = tid >> 2;         // 0..31
    const int ac = (tid & 3) * 8;    // 0,8,16,24 (halves)
    int rc = m0 + ar; if (rc >= cnt) rc = cnt - 1;
    const int tok = GATHER ? ltok[base + rc] : rc;
    const _Float16* asrc = A + (size_t)tok * KDIM + ac;
    _Float16* adst = &As[ar * 32 + ac];

    v8f cg00 = {}, cg01 = {}, cg10 = {}, cg11 = {};
    v8f cu00 = {}, cu01 = {}, cu10 = {}, cu11 = {};

    for (int k0 = 0; k0 < KDIM; k0 += 32) {
        __syncthreads();
        // A tile: async global->LDS b128 per thread
        copy16_g2l(asrc + k0, adst);
        // B tiles via VGPR (need f32->f16 conversion)
        stage_b_tile(wg + (size_t)k0 * NDIM, NDIM, Bg, tid);
        stage_b_tile(wu + (size_t)k0 * NDIM, NDIM, Bu, tid);
        if (k0 + 32 < KDIM) {
            __builtin_prefetch(wg + (size_t)(k0 + 32 + (tid & 15) * 2) * NDIM, 0, 1);
            __builtin_prefetch(wu + (size_t)(k0 + 32 + (tid & 15) * 2) * NDIM, 0, 1);
        }
        async_copy_wait();
        __syncthreads();

        // Load all fragments first, then issue WMMAs back-to-back
        v16h a0, a1, bg0, bu0, bg1, bu1;
        load_a_frag(As, lane, a0);
        load_a_frag(As + 16 * 32, lane, a1);
        const int nr0 = wave * 16 + (lane & 15);
        load_b_frag(Bg, nr0,      lane, bg0);
        load_b_frag(Bu, nr0,      lane, bu0);
        load_b_frag(Bg, nr0 + 64, lane, bg1);
        load_b_frag(Bu, nr0 + 64, lane, bu1);

        cg00 = WMMA_F16(a0, bg0, cg00);
        cg10 = WMMA_F16(a1, bg0, cg10);
        cu00 = WMMA_F16(a0, bu0, cu00);
        cu10 = WMMA_F16(a1, bu0, cu10);
        cg01 = WMMA_F16(a0, bg1, cg01);
        cg11 = WMMA_F16(a1, bg1, cg11);
        cu01 = WMMA_F16(a0, bu1, cu01);
        cu11 = WMMA_F16(a1, bu1, cu11);
    }

    // Epilogue: act = silu(g) * u  (C layout: VGPR j -> M = j + 8*half)
    const int nn0 = n0 + wave * 16 + (lane & 15);
#pragma unroll
    for (int j = 0; j < 8; ++j) {
        const int r0 = m0 + j + half * 8;
        const int r1 = r0 + 16;
        if (r0 < cnt) {
            _Float16* orow = Out + (size_t)(base + r0) * NDIM;
            orow[nn0]      = (_Float16)(silu_f(cg00[j]) * cu00[j]);
            orow[nn0 + 64] = (_Float16)(silu_f(cg01[j]) * cu01[j]);
        }
        if (r1 < cnt) {
            _Float16* orow = Out + (size_t)(base + r1) * NDIM;
            orow[nn0]      = (_Float16)(silu_f(cg10[j]) * cu10[j]);
            orow[nn0 + 64] = (_Float16)(silu_f(cg11[j]) * cu11[j]);
        }
    }
}

// ---------------------------------------------------------------------------
// 5) Down-projection WMMA GEMM.  Block tile M=32 x N=128, K-step 32.
// ---------------------------------------------------------------------------
template<bool EXPERT, bool F32OUT, int KDIM, int NDIM>
__global__ void down_kernel(const _Float16* __restrict__ A,   // [rows][KDIM]
                            const float* __restrict__ Wd,     // [e][KDIM][NDIM]
                            const int* __restrict__ counts,
                            const int* __restrict__ offs,
                            _Float16* __restrict__ OutH,      // f16 output
                            float* __restrict__ OutF) {       // f32 output
    const int e   = blockIdx.z;
    const int cnt = EXPERT ? counts[e] : TT;
    const int m0  = blockIdx.y * 32;
    if (m0 >= cnt) return;
    const int base = EXPERT ? offs[e] : 0;
    const int n0   = blockIdx.x * 128;

    const float* wd = Wd + (size_t)e * (size_t)KDIM * NDIM + n0;

    __shared__ _Float16 As[32 * 32];
    __shared__ _Float16 Bs[128 * 32];   // transposed [n][k]

    const int tid  = threadIdx.x;
    const int lane = tid & 31;
    const int wave = tid >> 5;
    const int half = lane >> 4;

    const int ar = tid >> 2;
    const int ac = (tid & 3) * 8;
    int rr = m0 + ar; if (rr >= cnt) rr = cnt - 1;
    const _Float16* asrc = A + (size_t)(base + rr) * KDIM + ac;
    _Float16* adst = &As[ar * 32 + ac];

    v8f c00 = {}, c01 = {}, c10 = {}, c11 = {};

    for (int k0 = 0; k0 < KDIM; k0 += 32) {
        __syncthreads();
        copy16_g2l(asrc + k0, adst);
        stage_b_tile(wd + (size_t)k0 * NDIM, NDIM, Bs, tid);
        if (k0 + 32 < KDIM)
            __builtin_prefetch(wd + (size_t)(k0 + 32 + (tid & 15) * 2) * NDIM, 0, 1);
        async_copy_wait();
        __syncthreads();

        v16h a0, a1, b0, b1;
        load_a_frag(As, lane, a0);
        load_a_frag(As + 16 * 32, lane, a1);
        const int nr0 = wave * 16 + (lane & 15);
        load_b_frag(Bs, nr0,      lane, b0);
        load_b_frag(Bs, nr0 + 64, lane, b1);

        c00 = WMMA_F16(a0, b0, c00);
        c10 = WMMA_F16(a1, b0, c10);
        c01 = WMMA_F16(a0, b1, c01);
        c11 = WMMA_F16(a1, b1, c11);
    }

    const int nn0 = n0 + wave * 16 + (lane & 15);
#pragma unroll
    for (int j = 0; j < 8; ++j) {
        const int r0 = m0 + j + half * 8;
        const int r1 = r0 + 16;
        if (r0 < cnt) {
            if (F32OUT) {
                float* orow = OutF + (size_t)r0 * NDIM;
                orow[nn0]      = c00[j];
                orow[nn0 + 64] = c01[j];
            } else {
                _Float16* orow = OutH + (size_t)(base + r0) * NDIM;
                orow[nn0]      = (_Float16)c00[j];
                orow[nn0 + 64] = (_Float16)c01[j];
            }
        }
        if (r1 < cnt) {
            if (F32OUT) {
                float* orow = OutF + (size_t)r1 * NDIM;
                orow[nn0]      = c10[j];
                orow[nn0 + 64] = c11[j];
            } else {
                _Float16* orow = OutH + (size_t)(base + r1) * NDIM;
                orow[nn0]      = (_Float16)c10[j];
                orow[nn0 + 64] = (_Float16)c11[j];
            }
        }
    }
}

// ---------------------------------------------------------------------------
// 6) Ordered (deterministic) top-k combine: routed[t] = sum_k w_k * y[row(t,k)]
//    One block per token; each thread handles 8 contiguous h (b128 loads).
// ---------------------------------------------------------------------------
__global__ void combine_kernel(const _Float16* __restrict__ y_all,
                               const int* __restrict__ row_of,
                               const float* __restrict__ topk_w,
                               float* __restrict__ routed) {
    __shared__ int   rs[TOPK];
    __shared__ float wsr[TOPK];
    const int t  = blockIdx.x;
    const int h0 = threadIdx.x * 8;
    if (threadIdx.x < TOPK) {
        rs[threadIdx.x]  = row_of[t * TOPK + threadIdx.x];
        wsr[threadIdx.x] = topk_w[t * TOPK + threadIdx.x];
    }
    __syncthreads();

    float acc[8];
#pragma unroll
    for (int j = 0; j < 8; ++j) acc[j] = 0.f;

    for (int k = 0; k < TOPK; ++k) {
        const _Float16* y = y_all + (size_t)rs[k] * HH + h0;
        const float w = wsr[k];
#pragma unroll
        for (int j = 0; j < 8; ++j) acc[j] += w * (float)y[j];
    }
    float* o = routed + (size_t)t * HH + h0;
#pragma unroll
    for (int j = 0; j < 8; ++j) o[j] = acc[j];
}

// ---------------------------------------------------------------------------
// Launch
// ---------------------------------------------------------------------------
extern "C" void kernel_launch(void* const* d_in, const int* in_sizes, int n_in,
                              void* d_out, int out_size, void* d_ws, size_t ws_size,
                              hipStream_t stream) {
    const float* x       = (const float*)d_in[0];   // [T,H]
    const float* gate_w  = (const float*)d_in[1];   // [E,H]
    const float* w_gate  = (const float*)d_in[2];   // [E,H,I]
    const float* w_up    = (const float*)d_in[3];   // [E,H,I]
    const float* w_down  = (const float*)d_in[4];   // [E,I,H]
    const float* sw_gate = (const float*)d_in[5];   // [H,IS]
    const float* sw_up   = (const float*)d_in[6];   // [H,IS]
    const float* sw_down = (const float*)d_in[7];   // [IS,H]
    (void)in_sizes; (void)n_in; (void)out_size; (void)ws_size;

    float* out_shared = (float*)d_out;                       // [T,H]
    float* out_routed = (float*)d_out + (size_t)TT * HH;     // [T,H]

    // Workspace layout
    char* ws = (char*)d_ws;
    size_t off = 0;
    _Float16* xh      = (_Float16*)(ws + off); off += (size_t)TT * HH * 2;       // 8 MB
    _Float16* act_buf = (_Float16*)(ws + off); off += (size_t)NROWS * II * 2;    // 12 MB
    _Float16* y_all   = (_Float16*)(ws + off); off += (size_t)NROWS * HH * 2;    // 48 MB
    _Float16* gs_buf  = (_Float16*)(ws + off); off += (size_t)TT * ISS * 2;      // 4 MB
    int*   topk_idx   = (int*)(ws + off);      off += (size_t)TT * TOPK * 4;
    float* topk_w     = (float*)(ws + off);    off += (size_t)TT * TOPK * 4;
    int*   counts     = (int*)(ws + off);      off += 256;
    int*   offs       = (int*)(ws + off);      off += 256;
    int*   ltok       = (int*)(ws + off);      off += (size_t)NROWS * 4;
    int*   row_of     = (int*)(ws + off);      off += (size_t)TT * TOPK * 4;

    // 1) x -> f16 (vectorized by 4)
    cvt_f32_f16_kernel<<<(TT * HH / 4 + 255) / 256, 256, 0, stream>>>(
        x, xh, TT * HH / 4);

    // 2) router
    router_kernel<<<TT, EE, 0, stream>>>(x, gate_w, topk_idx, topk_w);

    // 3) deterministic expert lists
    build_lists_kernel<<<1, EE, 0, stream>>>(topk_idx, counts, offs, ltok, row_of);

    // 4) expert gate/up + SiLU  (grid: N tiles x M tiles x experts)
    gateup_kernel<true, HH, II>
        <<<dim3(II / 128, TT / 32, EE), 128, 0, stream>>>(
            xh, w_gate, w_up, counts, offs, ltok, act_buf);

    // 5) expert down projection
    down_kernel<true, false, II, HH>
        <<<dim3(HH / 128, TT / 32, EE), 128, 0, stream>>>(
            act_buf, w_down, counts, offs, y_all, nullptr);

    // 6) combine (ordered, deterministic)
    combine_kernel<<<TT, HH / 8, 0, stream>>>(y_all, row_of, topk_w, out_routed);

    // 7) shared expert gate/up + SiLU
    gateup_kernel<false, HH, ISS>
        <<<dim3(ISS / 128, TT / 32, 1), 128, 0, stream>>>(
            xh, sw_gate, sw_up, nullptr, nullptr, nullptr, gs_buf);

    // 8) shared expert down projection -> f32 output
    down_kernel<false, true, ISS, HH>
        <<<dim3(HH / 128, TT / 32, 1), 128, 0, stream>>>(
            gs_buf, sw_down, nullptr, nullptr, nullptr, out_shared);
}
